// Attention_1795296330177
// MI455X (gfx1250) — matrix operand order repaired and checked by reading, not compiled, not run on hardware
//
#include <hip/hip_runtime.h>

#define B_  2
#define N_  2048
#define D_  1024
#define NH_ 16
#define HD_ 64
// MAXLEN == N_  -> rel_embed rows used are exactly rows 0..N_-1

typedef __attribute__((ext_vector_type(16))) _Float16 v16h;
typedef __attribute__((ext_vector_type(8)))  _Float16 v8h;
typedef __attribute__((ext_vector_type(8)))  float    v8f;

static __device__ __forceinline__ int lane_id() { return threadIdx.x & 31; }

#define WMMA_F16(a, b, c) \
  __builtin_amdgcn_wmma_f32_16x16x32_f16(false, (a), false, (b), (short)0, (c), false, false)

// ---------------- fragment loaders (CDNA5 wave32 16x16x32 f16 layouts) ----------------
// A (16x32, MxK): lane holds row = lane&15; halves e<8 -> K = kb+e, e>=8 -> K = kb+16+(e-8),
// kb = (lane&16) ? 8 : 0.   Source row-major, leading dim `ld` (halves, 16B-aligned rows).
static __device__ __forceinline__ v16h frag_a(const _Float16* base, int ld) {
  const int l = lane_id();
  const _Float16* p = base + (size_t)(l & 15) * ld + ((l & 16) ? 8 : 0);
  const v8h lo = *(const v8h*)p;
  const v8h hi = *(const v8h*)(p + 16);
  v16h f;
#pragma unroll
  for (int e = 0; e < 8; ++e) { f[e] = lo[e]; f[e + 8] = hi[e]; }
  return f;
}

// A fragment whose 16 rows are (row0 .. row0+15) of `mat`, clamped to >= 0 (shifted-Q load).
static __device__ __forceinline__ v16h frag_a_rowclamp(const _Float16* mat, int ld, int row0) {
  const int l = lane_id();
  int r = row0 + (l & 15);
  if (r < 0) r = 0;
  const _Float16* p = mat + (size_t)r * ld + ((l & 16) ? 8 : 0);
  const v8h lo = *(const v8h*)p;
  const v8h hi = *(const v8h*)(p + 16);
  v16h f;
#pragma unroll
  for (int e = 0; e < 8; ++e) { f[e] = lo[e]; f[e + 8] = hi[e]; }
  return f;
}

// B (32x16, KxN) for C = A * src^T: lane holds column n = lane&15 (a row of src);
// halves e<8 -> K = kb+e, e>=8 -> K = kb+16+(e-8), kb = (lane&16) ? 8 : 0.  Source row-major [N][K], contiguous over K.
static __device__ __forceinline__ v16h frag_b(const _Float16* base, int ld) {
  const int l = lane_id();
  const _Float16* p = base + (size_t)(l & 15) * ld + ((l & 16) ? 8 : 0);
  const v8h lo = *(const v8h*)p;
  const v8h hi = *(const v8h*)(p + 16);
  v16h f;
#pragma unroll
  for (int e = 0; e < 8; ++e) { f[e] = lo[e]; f[e + 8] = hi[e]; }
  return f;
}

// B fragment with source rows (row0 .. row0+15) clamped into [0, rmax] (rel-embed band edge).
static __device__ __forceinline__ v16h frag_b_rowclamp(const _Float16* mat, int ld, int row0, int rmax) {
  const int l = lane_id();
  int r = row0 + (l & 15);
  if (r < 0) r = 0;
  if (r > rmax) r = rmax;
  const _Float16* p = mat + (size_t)r * ld + ((l & 16) ? 8 : 0);
  const v8h lo = *(const v8h*)p;
  const v8h hi = *(const v8h*)(p + 16);
  v16h f;
#pragma unroll
  for (int e = 0; e < 8; ++e) { f[e] = lo[e]; f[e + 8] = hi[e]; }
  return f;
}

// B fragment from an LDS-staged tile, padded leading dim 40 halves (80 B rows, 16B aligned).
static __device__ __forceinline__ v16h frag_b_lds(const _Float16* base) {
  const int l = lane_id();
  const _Float16* p = base + (l & 15) * 40 + ((l & 16) ? 8 : 0);
  const v8h lo = *(const v8h*)p;
  const v8h hi = *(const v8h*)(p + 16);
  v16h f;
#pragma unroll
  for (int e = 0; e < 8; ++e) { f[e] = lo[e]; f[e + 8] = hi[e]; }
  return f;
}

// ---------------- async global->LDS staging of a 64(N) x 32(K) f16 weight tile ----------------
// Per wave: 2 x global_load_async_to_lds_b128 (32 lanes x 16B = 512B each); 4 waves cover 4KB.
// LDS layout: row n at byte offset n*80 (40-half padded stride).
static __device__ __forceinline__ void stage_b_tile(const _Float16* W, int K, int n0, int k0,
                                                    unsigned lds_base, int wave, int lane) {
#pragma unroll
  for (int s = 0; s < 2; ++s) {
    const int row = wave * 16 + s * 8 + (lane >> 2);   // 8 rows per instruction
    const int chunk = lane & 3;                        // 4 x 16B per 64B row
    const void* g = (const void*)(W + (size_t)(n0 + row) * K + k0 + chunk * 8);
    const unsigned lds = lds_base + (unsigned)(row * 80 + chunk * 16);
    asm volatile("global_load_async_to_lds_b128 %0, %1, off" :: "v"(lds), "v"(g) : "memory");
  }
}

// ---------------- f32 -> f16 convert ----------------
__global__ void cvt_f32_f16(const float* __restrict__ in, _Float16* __restrict__ out, int n) {
  int i = blockIdx.x * 256 + threadIdx.x;
  if (i < n) out[i] = (_Float16)in[i];
}

// ---------------- GEMM: C_f16[M][Nn] = A[M][K] * W[Nn][K]^T ----------------
// 4 waves, 64x64 block tile; B tile async-staged to LDS, double buffered.
// vmode==1: scatter output per-head-transposed for V: Vt[b][h][d][tok]
__global__ void __launch_bounds__(128)
gemm16(const _Float16* __restrict__ A, const _Float16* __restrict__ W,
       _Float16* __restrict__ O, int M, int Nn, int K, int vmode) {
  __shared__ _Float16 Bs[2][64 * 40];
  const int lane = threadIdx.x & 31;
  const int wave = threadIdx.x >> 5;
  const int m0 = blockIdx.x * 64 + wave * 16;
  const int n0 = blockIdx.y * 64;
  const unsigned bs0 = (unsigned)(uintptr_t)&Bs[0][0];
  const unsigned bs1 = (unsigned)(uintptr_t)&Bs[1][0];

  v8f acc[4];
#pragma unroll
  for (int t = 0; t < 4; ++t)
#pragma unroll
    for (int v = 0; v < 8; ++v) acc[t][v] = 0.0f;

  stage_b_tile(W, K, n0, 0, bs0, wave, lane);
  for (int k0 = 0; k0 < K; k0 += 32) {
    const int buf = (k0 >> 5) & 1;
    const bool more = (k0 + 32) < K;
    if (more) stage_b_tile(W, K, n0, k0 + 32, buf ? bs0 : bs1, wave, lane);
    if (more) asm volatile("s_wait_asynccnt 0x2" ::: "memory");   // current tile landed
    else      asm volatile("s_wait_asynccnt 0x0" ::: "memory");
    __syncthreads();                                              // publish to all waves
    __builtin_prefetch(A + (size_t)m0 * K + k0 + 128, 0, 3);
    v16h a = frag_a(A + (size_t)m0 * K + k0, K);
#pragma unroll
    for (int t = 0; t < 4; ++t) {
      v16h b = frag_b_lds(&Bs[buf][t * 16 * 40]);
      acc[t] = WMMA_F16(a, b, acc[t]);
    }
    __syncthreads();                                              // done reading buf
  }

  const int col = lane & 15;
  const int rbase = (lane & 16) ? 8 : 0;
#pragma unroll
  for (int t = 0; t < 4; ++t)
#pragma unroll
    for (int v = 0; v < 8; ++v) {
      const int m = m0 + rbase + v;
      const int nn = n0 + t * 16 + col;
      const _Float16 val = (_Float16)acc[t][v];
      if (vmode) {
        const int bb = m >> 11, tok = m & (N_ - 1), hh = nn >> 6, dd = nn & (HD_ - 1);
        O[(((size_t)bb * NH_ + hh) * HD_ + dd) * N_ + tok] = val;
      } else {
        O[(size_t)m * Nn + nn] = val;
      }
    }
}

// ---------------- final projection: out_f32 = A_f16 * W_f16^T + bias ----------------
__global__ void __launch_bounds__(128)
gemm_out(const _Float16* __restrict__ A, const _Float16* __restrict__ W,
         const float* __restrict__ bias, float* __restrict__ O, int M, int Nn, int K) {
  __shared__ _Float16 Bs[2][64 * 40];
  const int lane = threadIdx.x & 31;
  const int wave = threadIdx.x >> 5;
  const int m0 = blockIdx.x * 64 + wave * 16;
  const int n0 = blockIdx.y * 64;
  const unsigned bs0 = (unsigned)(uintptr_t)&Bs[0][0];
  const unsigned bs1 = (unsigned)(uintptr_t)&Bs[1][0];

  v8f acc[4];
#pragma unroll
  for (int t = 0; t < 4; ++t)
#pragma unroll
    for (int v = 0; v < 8; ++v) acc[t][v] = 0.0f;

  stage_b_tile(W, K, n0, 0, bs0, wave, lane);
  for (int k0 = 0; k0 < K; k0 += 32) {
    const int buf = (k0 >> 5) & 1;
    const bool more = (k0 + 32) < K;
    if (more) stage_b_tile(W, K, n0, k0 + 32, buf ? bs0 : bs1, wave, lane);
    if (more) asm volatile("s_wait_asynccnt 0x2" ::: "memory");
    else      asm volatile("s_wait_asynccnt 0x0" ::: "memory");
    __syncthreads();
    v16h a = frag_a(A + (size_t)m0 * K + k0, K);
#pragma unroll
    for (int t = 0; t < 4; ++t) {
      v16h b = frag_b_lds(&Bs[buf][t * 16 * 40]);
      acc[t] = WMMA_F16(a, b, acc[t]);
    }
    __syncthreads();
  }

  const int col = lane & 15;
  const int rbase = (lane & 16) ? 8 : 0;
#pragma unroll
  for (int t = 0; t < 4; ++t)
#pragma unroll
    for (int v = 0; v < 8; ++v) {
      const int m = m0 + rbase + v;
      const int nn = n0 + t * 16 + col;
      O[(size_t)m * Nn + nn] = acc[t][v] + bias[nn];
    }
}

// ---------------- flash attention with music-transformer skewed rel bias ----------------
// one wave per (b, h, 16-row tile); 32-key chunks; causal.
// srel[i,j] = (Q' E^T)[i, N + j - i] for j<i (Q'[i]=q_{i-1}); 0 at j==i; masked for j>i.
__global__ void __launch_bounds__(32)
attn(const _Float16* __restrict__ Q, const _Float16* __restrict__ Km,
     const _Float16* __restrict__ Vt, const _Float16* __restrict__ E,
     _Float16* __restrict__ O) {
  __shared__ float    Gs[16][48];   // srel band tile
  __shared__ _Float16 Ps[16][32];   // probabilities -> A fragment

  const int i0 = blockIdx.x * 16;
  const int h  = blockIdx.y;
  const int b  = blockIdx.z;
  const int l  = lane_id();
  const int col = l & 15;
  const int rbase = (l & 16) ? 8 : 0;

  const _Float16* Qb = Q  + ((size_t)b * N_) * D_ + h * HD_;
  const _Float16* Kb = Km + ((size_t)b * N_) * D_ + h * HD_;
  const _Float16* Vb = Vt + ((size_t)(b * NH_ + h)) * HD_ * N_;   // [HD][N]
  const _Float16* Eb = E + h * HD_;                                // rel_embed rows, ld = D_

  // Q rows i0..i0+15 (K split 0..31 / 32..63), and shifted rows i0-1..i0+14 for srel
  const v16h qa0 = frag_a(Qb + (size_t)i0 * D_,      D_);
  const v16h qa1 = frag_a(Qb + (size_t)i0 * D_ + 32, D_);
  const v16h pa0 = frag_a_rowclamp(Qb,      D_, i0 - 1);
  const v16h pa1 = frag_a_rowclamp(Qb + 32, D_, i0 - 1);

  v8f o[4];
  float mrow[8], lrow[8];
#pragma unroll
  for (int t = 0; t < 4; ++t)
#pragma unroll
    for (int v = 0; v < 8; ++v) o[t][v] = 0.0f;
#pragma unroll
  for (int v = 0; v < 8; ++v) { mrow[v] = -3.0e38f; lrow[v] = 0.0f; }

  const int nchunk = i0 / 32 + 1;   // covers all j <= i0+15
  for (int ch = 0; ch < nchunk; ++ch) {
    const int j0 = ch * 32;
    // ---- scores S = Q K^T (16 x 32) ----
    v8f slo, shi;
#pragma unroll
    for (int v = 0; v < 8; ++v) { slo[v] = 0.0f; shi[v] = 0.0f; }
    slo = WMMA_F16(qa0, frag_b(Kb + (size_t)j0 * D_,             D_), slo);
    slo = WMMA_F16(qa1, frag_b(Kb + (size_t)j0 * D_ + 32,        D_), slo);
    shi = WMMA_F16(qa0, frag_b(Kb + (size_t)(j0 + 16) * D_,      D_), shi);
    shi = WMMA_F16(qa1, frag_b(Kb + (size_t)(j0 + 16) * D_ + 32, D_), shi);

    // ---- srel band G = Q' E^T over columns c0 .. c0+47 ----
    const int c0 = N_ + j0 - i0 - 15;
#pragma unroll
    for (int t = 0; t < 3; ++t) {
      v8f g;
#pragma unroll
      for (int v = 0; v < 8; ++v) g[v] = 0.0f;
      g = WMMA_F16(pa0, frag_b_rowclamp(Eb,      D_, c0 + t * 16, N_ - 1), g);
      g = WMMA_F16(pa1, frag_b_rowclamp(Eb + 32, D_, c0 + t * 16, N_ - 1), g);
#pragma unroll
      for (int v = 0; v < 8; ++v) Gs[rbase + v][t * 16 + col] = g[v];
    }
    asm volatile("s_wait_dscnt 0" ::: "memory");

    // ---- assemble scores + online softmax ----
#pragma unroll
    for (int v = 0; v < 8; ++v) {
      const int r = rbase + v;
      const int i = i0 + r;
      float s0, s1;
      {
        const int j = j0 + col;
        if (j > i) s0 = -3.0e38f;
        else {
          const float sr = (j == i) ? 0.0f : Gs[r][15 + (j - j0) - r];
          s0 = (slo[v] + sr) * 0.125f;   // 1/sqrt(HD)
        }
      }
      {
        const int j = j0 + 16 + col;
        if (j > i) s1 = -3.0e38f;
        else {
          const float sr = (j == i) ? 0.0f : Gs[r][15 + (j - j0) - r];
          s1 = (shi[v] + sr) * 0.125f;
        }
      }
      float rm = fmaxf(s0, s1);
      rm = fmaxf(rm, __shfl_xor(rm, 1, 32));
      rm = fmaxf(rm, __shfl_xor(rm, 2, 32));
      rm = fmaxf(rm, __shfl_xor(rm, 4, 32));
      rm = fmaxf(rm, __shfl_xor(rm, 8, 32));
      const float mnew = fmaxf(mrow[v], rm);
      const float p0 = __expf(s0 - mnew);
      const float p1 = __expf(s1 - mnew);
      float ps = p0 + p1;
      ps += __shfl_xor(ps, 1, 32);
      ps += __shfl_xor(ps, 2, 32);
      ps += __shfl_xor(ps, 4, 32);
      ps += __shfl_xor(ps, 8, 32);
      const float alpha = __expf(mrow[v] - mnew);
      lrow[v] = lrow[v] * alpha + ps;
      mrow[v] = mnew;
#pragma unroll
      for (int t = 0; t < 4; ++t) o[t][v] *= alpha;
      Ps[r][col]      = (_Float16)p0;
      Ps[r][col + 16] = (_Float16)p1;
    }
    asm volatile("s_wait_dscnt 0" ::: "memory");

    // ---- O += P V : P is a 16x32 A fragment; V^T is contiguous in Vt[d][j] ----
    const v16h pf = frag_a(&Ps[0][0], 32);
#pragma unroll
    for (int t = 0; t < 4; ++t) {
      v16h vb = frag_b(Vb + (size_t)(t * 16) * N_ + j0, N_);
      o[t] = WMMA_F16(pf, vb, o[t]);
    }
  }

  // ---- epilogue: normalize, store f16 [B][N][D] ----
#pragma unroll
  for (int v = 0; v < 8; ++v) {
    const int i = i0 + rbase + v;
    const float inv = 1.0f / lrow[v];
#pragma unroll
    for (int t = 0; t < 4; ++t)
      O[((size_t)b * N_ + i) * D_ + h * HD_ + t * 16 + col] = (_Float16)(o[t][v] * inv);
  }
}

// ---------------- host ----------------
extern "C" void kernel_launch(void* const* d_in, const int* in_sizes, int n_in,
                              void* d_out, int out_size, void* d_ws, size_t ws_size,
                              hipStream_t stream) {
  (void)in_sizes; (void)n_in; (void)out_size; (void)ws_size;
  const float* x  = (const float*)d_in[0];
  const float* Wq = (const float*)d_in[1];
  const float* Wk = (const float*)d_in[2];
  const float* Wv = (const float*)d_in[3];
  const float* Wp = (const float*)d_in[4];
  const float* bp = (const float*)d_in[5];
  const float* re = (const float*)d_in[6];
  float* out = (float*)d_out;

  const int TOK = B_ * N_;                 // 4096
  char* ws = (char*)d_ws;
  size_t off = 0;
  auto carve = [&](size_t bytes) { void* p = ws + off; off += bytes; return p; };
  _Float16* Xh  = (_Float16*)carve((size_t)TOK * D_ * 2);   // 8 MiB
  _Float16* Wqh = (_Float16*)carve((size_t)D_ * D_ * 2);    // 2 MiB
  _Float16* Wkh = (_Float16*)carve((size_t)D_ * D_ * 2);
  _Float16* Wvh = (_Float16*)carve((size_t)D_ * D_ * 2);
  _Float16* Wph = (_Float16*)carve((size_t)D_ * D_ * 2);
  _Float16* Eh  = (_Float16*)carve((size_t)N_ * D_ * 2);    // 4 MiB
  _Float16* Qh  = (_Float16*)carve((size_t)TOK * D_ * 2);   // 8 MiB
  _Float16* Kh  = (_Float16*)carve((size_t)TOK * D_ * 2);
  _Float16* Vh  = (_Float16*)carve((size_t)TOK * D_ * 2);   // per-head transposed
  _Float16* AOh = (_Float16*)carve((size_t)TOK * D_ * 2);   // total ~52 MiB

  cvt_f32_f16<<<(TOK * D_ + 255) / 256, 256, 0, stream>>>(x,  Xh,  TOK * D_);
  cvt_f32_f16<<<(D_ * D_ + 255) / 256, 256, 0, stream>>>(Wq, Wqh, D_ * D_);
  cvt_f32_f16<<<(D_ * D_ + 255) / 256, 256, 0, stream>>>(Wk, Wkh, D_ * D_);
  cvt_f32_f16<<<(D_ * D_ + 255) / 256, 256, 0, stream>>>(Wv, Wvh, D_ * D_);
  cvt_f32_f16<<<(D_ * D_ + 255) / 256, 256, 0, stream>>>(Wp, Wph, D_ * D_);
  cvt_f32_f16<<<(N_ * D_ + 255) / 256, 256, 0, stream>>>(re, Eh,  N_ * D_);

  dim3 gg(TOK / 64, D_ / 64);
  gemm16<<<gg, 128, 0, stream>>>(Xh, Wqh, Qh, TOK, D_, D_, 0);
  gemm16<<<gg, 128, 0, stream>>>(Xh, Wkh, Kh, TOK, D_, D_, 0);
  gemm16<<<gg, 128, 0, stream>>>(Xh, Wvh, Vh, TOK, D_, D_, 1);

  dim3 ga(N_ / 16, NH_, B_);
  attn<<<ga, 32, 0, stream>>>(Qh, Kh, Vh, Eh, AOh);

  gemm_out<<<gg, 128, 0, stream>>>(AOh, Wph, bp, out, TOK, D_, D_);
}
